// MultiHeadedAttention_85736137162970
// MI455X (gfx1250) — compile-verified, matrix operand order checked by reading, not run on hardware
//
#include <hip/hip_runtime.h>

// Problem constants (fixed by the reference).
#define B_   2
#define S_   2048
#define D_   1024
#define H_   16
#define DK_  64
#define M_   (B_ * S_)   // 4096 rows of the projection GEMMs
#define BH_  (B_ * H_)   // 32 (batch*heads)

typedef __attribute__((ext_vector_type(16))) __bf16 v16bf;
typedef __attribute__((ext_vector_type(8)))  __bf16 v8bf;
typedef __attribute__((ext_vector_type(8)))  float  v8f;
typedef __attribute__((ext_vector_type(4)))  float  v4f;

// ---- fragment loader --------------------------------------------------------
// 16-bit A/B fragment per-lane pattern (ISA 7.12.2): with p already offset by
// hi*8, the lane holds contraction elements p[0..7] and p[16..23].
static __device__ __forceinline__ v16bf load_frag_bf16(const __bf16* p) {
  const v8bf lo = *(const v8bf*)(p);
  const v8bf hv = *(const v8bf*)(p + 16);
  v16bf r;
#pragma unroll
  for (int j = 0; j < 8; ++j) { r[j] = lo[j]; r[j + 8] = hv[j]; }
  return r;
}

static __device__ __forceinline__ v8f wmma_bf16(v16bf a, v16bf b, v8f c) {
  return __builtin_amdgcn_wmma_f32_16x16x32_bf16(false, a, false, b,
                                                 (short)0, c, false, false);
}

// ---- stage 0: one-shot f32 -> bf16 conversion (each element exactly once) --
__global__ __launch_bounds__(256)
void cvt_f32_to_bf16_kernel(const float* __restrict__ in,
                            __bf16* __restrict__ out, int n8) {
  const int i = blockIdx.x * blockDim.x + threadIdx.x;
  if (i < n8) {
    const v8f x = *(const v8f*)(in + (size_t)i * 8);
    v8bf y;
#pragma unroll
    for (int j = 0; j < 8; ++j) y[j] = (__bf16)x[j];
    *(v8bf*)(out + (size_t)i * 8) = y;
  }
}

// ---- stage 1: fused Q/K/V projection, 64x64 tile per wave ------------------
// All operands pre-converted bf16 -> inner loop is pure b128 loads + WMMA.
// gridDim.z selects Q(0) / K(1) / V(2).
//  z<2 ("swapped"):  C = W(A) x x^T(B)  -> M=feature, N=row.  8 consecutive
//       features per lane -> packed v8bf stores into head-major [BH][S][64].
//       Q additionally folds in the 1/sqrt(dk) score scale.
//  z==2 ("normal"):  C = x(A) x W^T(B) -> M=row, N=feature.  8 consecutive
//       rows per lane -> packed v8bf stores into transposed [BH][64][S].
__global__ __launch_bounds__(32)
void qkv_proj_kernel(const __bf16* __restrict__ Xq, const __bf16* __restrict__ Xk,
                     const __bf16* __restrict__ Xv,
                     const __bf16* __restrict__ Wq, const __bf16* __restrict__ Wk,
                     const __bf16* __restrict__ Wv,
                     const float* __restrict__ Bq, const float* __restrict__ Bk,
                     const float* __restrict__ Bv,
                     __bf16* __restrict__ Qb, __bf16* __restrict__ Kb,
                     __bf16* __restrict__ Vt) {
  const int z = blockIdx.z;
  const __bf16* X    = (z == 0) ? Xq : (z == 1) ? Xk : Xv;
  const __bf16* W    = (z == 0) ? Wq : (z == 1) ? Wk : Wv;
  const float*  bias = (z == 0) ? Bq : (z == 1) ? Bk : Bv;

  const int lane = threadIdx.x;
  const int m    = lane & 15;
  const int hi   = lane >> 4;
  const int f0   = blockIdx.x * 64;   // feature tile base
  const int r0   = blockIdx.y * 64;   // (b*s) row tile base

  const __bf16* wbase = W + (size_t)(f0 + m) * D_;
  const __bf16* xbase = X + (size_t)(r0 + m) * D_;

  v8f acc[4][4];
  if (z < 2) {   // bias along M (features)
#pragma unroll
    for (int mt = 0; mt < 4; ++mt)
#pragma unroll
      for (int v = 0; v < 8; ++v) {
        const float bval = bias[f0 + mt * 16 + 8 * hi + v];
#pragma unroll
        for (int nt = 0; nt < 4; ++nt) acc[mt][nt][v] = bval;
      }
  } else {       // bias along N (features)
#pragma unroll
    for (int nt = 0; nt < 4; ++nt) {
      const float bval = bias[f0 + nt * 16 + m];
#pragma unroll
      for (int mt = 0; mt < 4; ++mt)
#pragma unroll
        for (int v = 0; v < 8; ++v) acc[mt][nt][v] = bval;
    }
  }

  if (z < 2) {
    for (int k0 = 0; k0 < D_; k0 += 32) {
      v16bf af[4];
#pragma unroll
      for (int mt = 0; mt < 4; ++mt)
        af[mt] = load_frag_bf16(wbase + (size_t)mt * 16 * D_ + k0 + hi * 8);
#pragma unroll
      for (int nt = 0; nt < 4; ++nt) {
        const v16bf b = load_frag_bf16(xbase + (size_t)nt * 16 * D_ + k0 + hi * 8);
#pragma unroll
        for (int mt = 0; mt < 4; ++mt) acc[mt][nt] = wmma_bf16(af[mt], b, acc[mt][nt]);
      }
    }
    __bf16* out = (z == 0) ? Qb : Kb;
    const float scale = (z == 0) ? 0.125f : 1.0f;   // fold 1/sqrt(64) into Q
#pragma unroll
    for (int mt = 0; mt < 4; ++mt) {
      const int f   = f0 + mt * 16 + 8 * hi;  // 8 consecutive features
      const int h   = f >> 6;
      const int dk0 = f & (DK_ - 1);
#pragma unroll
      for (int nt = 0; nt < 4; ++nt) {
        const int r  = r0 + nt * 16 + m;
        const int b_ = r >> 11;
        const int s_ = r & (S_ - 1);
        v8bf pk;
#pragma unroll
        for (int v = 0; v < 8; ++v) pk[v] = (__bf16)(acc[mt][nt][v] * scale);
        *(v8bf*)(out + ((size_t)(b_ * H_ + h) * S_ + s_) * DK_ + dk0) = pk;
      }
    }
  } else {
    for (int k0 = 0; k0 < D_; k0 += 32) {
      v16bf af[4];
#pragma unroll
      for (int mt = 0; mt < 4; ++mt)
        af[mt] = load_frag_bf16(xbase + (size_t)mt * 16 * D_ + k0 + hi * 8);
#pragma unroll
      for (int nt = 0; nt < 4; ++nt) {
        const v16bf b = load_frag_bf16(wbase + (size_t)nt * 16 * D_ + k0 + hi * 8);
#pragma unroll
        for (int mt = 0; mt < 4; ++mt) acc[mt][nt] = wmma_bf16(af[mt], b, acc[mt][nt]);
      }
    }
#pragma unroll
    for (int nt = 0; nt < 4; ++nt) {
      const int f  = f0 + nt * 16 + m;
      const int h  = f >> 6;
      const int dk = f & (DK_ - 1);
#pragma unroll
      for (int mt = 0; mt < 4; ++mt) {
        const int rb = r0 + mt * 16 + 8 * hi;   // 8 consecutive rows
        const int b_ = rb >> 11;
        const int s0 = rb & (S_ - 1);
        v8bf pk;
#pragma unroll
        for (int v = 0; v < 8; ++v) pk[v] = (__bf16)acc[mt][nt][v];
        *(v8bf*)(Vt + ((size_t)(b_ * H_ + h) * DK_ + dk) * S_ + s0) = pk;
      }
    }
  }
}

// ---- stage 2: flash attention, transposed-score trick ----------------------
// One wave = 32 queries of one (b,h). Scores computed as S^T = K(A) x Q^T(B):
// lane index = query, VGPR index = key. Per-query softmax needs only an
// in-lane tree + one shfl_xor(16). The probability tile in registers is
// *already* the B-fragment layout for ctx^T = V^T(A) x P^T(B): no LDS at all.
__global__ __launch_bounds__(32)
void attn_kernel(const __bf16* __restrict__ Qb, const __bf16* __restrict__ Kb,
                 const __bf16* __restrict__ Vt, __bf16* __restrict__ Ctx) {
  const int lane = threadIdx.x;
  const int m    = lane & 15;
  const int hi   = lane >> 4;
  const int q0   = blockIdx.x * 32;
  const int bh   = blockIdx.y;

  // Q as B-operand: column n = query q0 + mt*16 + n (contiguous over d).
  v16bf qb[2][2];
#pragma unroll
  for (int mt = 0; mt < 2; ++mt) {
    const __bf16* qrow = Qb + ((size_t)bh * S_ + q0 + mt * 16 + m) * DK_;
    qb[mt][0] = load_frag_bf16(qrow + hi * 8);
    qb[mt][1] = load_frag_bf16(qrow + 32 + hi * 8);
  }

  v8f acc[2][4];
#pragma unroll
  for (int mt = 0; mt < 2; ++mt)
#pragma unroll
    for (int nt = 0; nt < 4; ++nt)
#pragma unroll
      for (int v = 0; v < 8; ++v) acc[mt][nt][v] = 0.0f;

  float mx[2]   = {-3.402823466e38f, -3.402823466e38f};
  float lsum[2] = {0.0f, 0.0f};

  for (int j0 = 0; j0 < S_; j0 += 32) {
    // Prefetch next strip's K/V lines (global_prefetch_b8 path).
    if (j0 + 32 < S_) {
      __builtin_prefetch(Kb + ((size_t)bh * S_ + j0 + 32 + lane) * DK_, 0, 3);
      __builtin_prefetch(Vt + ((size_t)bh * DK_ + 2 * lane) * S_ + j0 + 32, 0, 3);
    }

    // -- S^T tiles: (key, query); st[mt][kt][v] = score(key kt*16+v+8*hi, query) --
    v8f st[2][2];
    const v8f z8 = {};
#pragma unroll
    for (int kt = 0; kt < 2; ++kt) {
      const __bf16* krow = Kb + ((size_t)bh * S_ + j0 + kt * 16 + m) * DK_;
      const v16bf ka0 = load_frag_bf16(krow + hi * 8);
      const v16bf ka1 = load_frag_bf16(krow + 32 + hi * 8);
#pragma unroll
      for (int mt = 0; mt < 2; ++mt) {
        v8f s = wmma_bf16(ka0, qb[mt][0], z8);
        st[mt][kt] = wmma_bf16(ka1, qb[mt][1], s);
      }
    }

    // -- exact per-query online softmax: in-lane tree + one cross-half xor --
    v16bf pb[2];
    float corr[2];
#pragma unroll
    for (int mt = 0; mt < 2; ++mt) {
      float vmax = st[mt][0][0];
#pragma unroll
      for (int v = 1; v < 8; ++v) vmax = fmaxf(vmax, st[mt][0][v]);
#pragma unroll
      for (int v = 0; v < 8; ++v) vmax = fmaxf(vmax, st[mt][1][v]);
      vmax = fmaxf(vmax, __shfl_xor(vmax, 16, 32));   // merge other 16 keys
      const float mnew = fmaxf(mx[mt], vmax);
      const float cf   = __expf(mx[mt] - mnew);
      float rsum = 0.0f;
#pragma unroll
      for (int v = 0; v < 8; ++v) {
        const float p0 = __expf(st[mt][0][v] - mnew);
        const float p1 = __expf(st[mt][1][v] - mnew);
        pb[mt][v]     = (__bf16)p0;   // B-frag: contraction key 8*hi+v
        pb[mt][v + 8] = (__bf16)p1;   // B-frag: contraction key 16+8*hi+v
        rsum += p0 + p1;
      }
      rsum += __shfl_xor(rsum, 16, 32);
      lsum[mt] = lsum[mt] * cf + rsum;
      mx[mt]   = mnew;
      corr[mt] = cf;
    }
#pragma unroll
    for (int mt = 0; mt < 2; ++mt)
#pragma unroll
      for (int nt = 0; nt < 4; ++nt)
#pragma unroll
        for (int v = 0; v < 8; ++v) acc[mt][nt][v] *= corr[mt];

    // -- ctx^T += V^T(16 dims x 32 keys) x P^T(32 keys x 16 queries) --
#pragma unroll
    for (int nt = 0; nt < 4; ++nt) {
      const __bf16* vrow = Vt + ((size_t)bh * DK_ + nt * 16 + m) * S_ + j0;
      const v16bf va = load_frag_bf16(vrow + hi * 8);
#pragma unroll
      for (int mt = 0; mt < 2; ++mt)
        acc[mt][nt] = wmma_bf16(va, pb[mt], acc[mt][nt]);
    }
  }

  // -- normalize; 8 consecutive dims per lane -> packed v8bf stores ----------
  const int b_ = bh >> 4;
  const int h  = bh & 15;
#pragma unroll
  for (int mt = 0; mt < 2; ++mt) {
    const float rl = 1.0f / lsum[mt];
    const int  s_  = q0 + mt * 16 + m;   // query = lane column
    __bf16* crow = Ctx + ((size_t)(b_ * S_ + s_)) * D_ + h * DK_ + 8 * hi;
#pragma unroll
    for (int nt = 0; nt < 4; ++nt) {
      v8bf pk;
#pragma unroll
      for (int v = 0; v < 8; ++v) pk[v] = (__bf16)(acc[mt][nt][v] * rl);
      *(v8bf*)(crow + nt * 16) = pk;
    }
  }
}

// ---- stage 3: output projection (swapped orientation, f32 out) -------------
__global__ __launch_bounds__(32)
void out_proj_kernel(const __bf16* __restrict__ Xc, const __bf16* __restrict__ W,
                     const float* __restrict__ bias, float* __restrict__ out) {
  const int lane = threadIdx.x;
  const int m    = lane & 15;
  const int hi   = lane >> 4;
  const int f0   = blockIdx.x * 64;   // output column tile
  const int r0   = blockIdx.y * 64;   // row tile

  const __bf16* wbase = W  + (size_t)(f0 + m) * D_;
  const __bf16* xbase = Xc + (size_t)(r0 + m) * D_;

  v8f acc[4][4];
#pragma unroll
  for (int mt = 0; mt < 4; ++mt)
#pragma unroll
    for (int v = 0; v < 8; ++v) {
      const float bval = bias[f0 + mt * 16 + 8 * hi + v];
#pragma unroll
      for (int nt = 0; nt < 4; ++nt) acc[mt][nt][v] = bval;
    }

  for (int k0 = 0; k0 < D_; k0 += 32) {
    v16bf af[4];
#pragma unroll
    for (int mt = 0; mt < 4; ++mt)
      af[mt] = load_frag_bf16(wbase + (size_t)mt * 16 * D_ + k0 + hi * 8);
#pragma unroll
    for (int nt = 0; nt < 4; ++nt) {
      const v16bf b = load_frag_bf16(xbase + (size_t)nt * 16 * D_ + k0 + hi * 8);
#pragma unroll
      for (int mt = 0; mt < 4; ++mt) acc[mt][nt] = wmma_bf16(af[mt], b, acc[mt][nt]);
    }
  }

#pragma unroll
  for (int mt = 0; mt < 4; ++mt) {
    const int c0 = f0 + mt * 16 + 8 * hi;   // 8 consecutive f32 columns
#pragma unroll
    for (int nt = 0; nt < 4; ++nt) {
      const int row = r0 + nt * 16 + m;
      float* p = out + (size_t)row * D_ + c0;
      v4f lo, hv;
#pragma unroll
      for (int v = 0; v < 4; ++v) { lo[v] = acc[mt][nt][v]; hv[v] = acc[mt][nt][v + 4]; }
      *(v4f*)(p)     = lo;
      *(v4f*)(p + 4) = hv;
    }
  }
}

// ---- launcher ---------------------------------------------------------------
extern "C" void kernel_launch(void* const* d_in, const int* in_sizes, int n_in,
                              void* d_out, int out_size, void* d_ws, size_t ws_size,
                              hipStream_t stream) {
  (void)in_sizes; (void)n_in; (void)out_size; (void)ws_size;

  const float* q  = (const float*)d_in[0];
  const float* k  = (const float*)d_in[1];
  const float* v  = (const float*)d_in[2];
  const float* Wq = (const float*)d_in[3];
  const float* bq = (const float*)d_in[4];
  const float* Wk = (const float*)d_in[5];
  const float* bk = (const float*)d_in[6];
  const float* Wv = (const float*)d_in[7];
  const float* bv = (const float*)d_in[8];
  const float* Wo = (const float*)d_in[9];
  const float* bo = (const float*)d_in[10];

  // Workspace layout (64 MB total):
  //   4 x [B*S*D] bf16:  Qb, Kb, Vt, Ctx                (4 x 8 MB)
  //   3 x [B*S*D] bf16:  xq, xk, xv converted activations (3 x 8 MB)
  //   4 x [D*D]   bf16:  Wq, Wk, Wv, Wo converted weights (4 x 2 MB)
  char* ws = (char*)d_ws;
  const size_t act_b = (size_t)M_ * D_ * sizeof(__bf16);  // 8 MB
  const size_t wgt_b = (size_t)D_ * D_ * sizeof(__bf16);  // 2 MB
  __bf16* Qb   = (__bf16*)(ws + 0 * act_b);
  __bf16* Kb   = (__bf16*)(ws + 1 * act_b);
  __bf16* Vt   = (__bf16*)(ws + 2 * act_b);
  __bf16* Ctx  = (__bf16*)(ws + 3 * act_b);
  __bf16* Xqb  = (__bf16*)(ws + 4 * act_b);
  __bf16* Xkb  = (__bf16*)(ws + 5 * act_b);
  __bf16* Xvb  = (__bf16*)(ws + 6 * act_b);
  __bf16* Wqb  = (__bf16*)(ws + 7 * act_b + 0 * wgt_b);
  __bf16* Wkb  = (__bf16*)(ws + 7 * act_b + 1 * wgt_b);
  __bf16* Wvb  = (__bf16*)(ws + 7 * act_b + 2 * wgt_b);
  __bf16* Wob  = (__bf16*)(ws + 7 * act_b + 3 * wgt_b);

  const int act_n8 = (M_ * D_) / 8;   // 524288
  const int wgt_n8 = (D_ * D_) / 8;   // 131072
  const dim3 cblk(256, 1, 1);
  const dim3 cga((act_n8 + 255) / 256, 1, 1);
  const dim3 cgw((wgt_n8 + 255) / 256, 1, 1);
  cvt_f32_to_bf16_kernel<<<cga, cblk, 0, stream>>>(q,  Xqb, act_n8);
  cvt_f32_to_bf16_kernel<<<cga, cblk, 0, stream>>>(k,  Xkb, act_n8);
  cvt_f32_to_bf16_kernel<<<cga, cblk, 0, stream>>>(v,  Xvb, act_n8);
  cvt_f32_to_bf16_kernel<<<cgw, cblk, 0, stream>>>(Wq, Wqb, wgt_n8);
  cvt_f32_to_bf16_kernel<<<cgw, cblk, 0, stream>>>(Wk, Wkb, wgt_n8);
  cvt_f32_to_bf16_kernel<<<cgw, cblk, 0, stream>>>(Wv, Wvb, wgt_n8);
  cvt_f32_to_bf16_kernel<<<cgw, cblk, 0, stream>>>(Wo, Wob, wgt_n8);

  const dim3 blk(32, 1, 1);
  const dim3 gproj(D_ / 64, M_ / 64, 3);   // (16, 64, 3): Q,K,V concurrently
  const dim3 gattn(S_ / 32, BH_, 1);       // (64, 32): 32 queries per wave
  const dim3 gout (D_ / 64, M_ / 64, 1);   // (16, 64)

  qkv_proj_kernel<<<gproj, blk, 0, stream>>>(Xqb, Xkb, Xvb, Wqb, Wkb, Wvb,
                                             bq, bk, bv, Qb, Kb, Vt);
  attn_kernel<<<gattn, blk, 0, stream>>>(Qb, Kb, Vt, Ctx);
  out_proj_kernel<<<gout, blk, 0, stream>>>(Ctx, Wob, bo, (float*)d_out);
}
